// GSAFilter_66872640798820
// MI455X (gfx1250) — compile-verified, matrix operand order checked by reading, not run on hardware
//
#include <hip/hip_runtime.h>
#include <hip/hip_bf16.h>
#include <math.h>

typedef __attribute__((ext_vector_type(16))) _Float16 v16h;
typedef __attribute__((ext_vector_type(8)))  float    v8f;
typedef _Float16 half_t;

// Problem constants
constexpr int kB = 8, kT = 1024, kH = 8;
constexpr int kDM = 512, kDA = 256, kDP = 128;
constexpr int kDK = 64, kDAh = 32, kDPh = 16;
constexpr int kNT = kB * kT;            // 8192 tokens
constexpr int kM1 = 8, kM2 = 8;

// ---------------------------------------------------------------------------
// WMMA A/B-operand register layout helpers (16-bit, 16x32 tile, wave32):
// lane L holds row m = L&15; half h = L>>4 selects the K subset.
// VGPR pair v (v=0..7) holds K = 16*(v>>2) + 8*h + 2*(v&3) + {0,1}.
// B operand is loaded identically from the [N,K] row-major matrix (i.e. B^T),
// so D[m,n] = sum_k A[m,k] * W[n,k].
// C/D layout: vgpr r -> m = r + 8*h, n = lane&15.
// ---------------------------------------------------------------------------
__device__ __forceinline__ int koff_pair(int v, int hh) {
  return ((v >> 2) << 4) + (hh << 3) + ((v & 3) << 1);
}

__device__ __forceinline__ v16h load_tile_f16(const half_t* __restrict__ p, int hh, bool ok) {
  v16h r;
#pragma unroll
  for (int v = 0; v < 8; ++v) {
    int k = koff_pair(v, hh);
    r[2 * v]     = ok ? p[k]     : (half_t)0.0f;
    r[2 * v + 1] = ok ? p[k + 1] : (half_t)0.0f;
  }
  return r;
}

__device__ __forceinline__ v16h load_tile_f32cvt(const float* __restrict__ p, int hh) {
  v16h r;
#pragma unroll
  for (int v = 0; v < 8; ++v) {
    int k = koff_pair(v, hh);
    r[2 * v]     = (half_t)p[k];
    r[2 * v + 1] = (half_t)p[k + 1];
  }
  return r;
}

__device__ __forceinline__ v8f zero8() {
  v8f z = {0.f, 0.f, 0.f, 0.f, 0.f, 0.f, 0.f, 0.f};
  return z;
}

__device__ __forceinline__ v8f wmma_f16(v16h a, v16h b, v8f c) {
  return __builtin_amdgcn_wmma_f32_16x16x32_f16(false, a, false, b, (short)0, c, false, false);
}

// ---------------------------------------------------------------------------
// Elementwise prep kernels
// ---------------------------------------------------------------------------
__global__ void cvt_f16_kernel(const float* __restrict__ s, half_t* __restrict__ d, int n) {
  int i = blockIdx.x * blockDim.x + threadIdx.x;
  int stride = gridDim.x * blockDim.x;
  for (; i < n; i += stride) d[i] = (half_t)s[i];
}

// Wq/Wk/Wv: [H,DK,DM] flat; mask = g[k, m%64]
__global__ void mask_qkv_kernel(const float* __restrict__ W, const int* __restrict__ g,
                                half_t* __restrict__ d) {
  int i = blockIdx.x * blockDim.x + threadIdx.x;
  if (i >= kH * kDK * kDM) return;
  int m = i & (kDM - 1);
  int k = (i >> 9) & (kDK - 1);
  d[i] = (half_t)(g[(k << 6) + (m & 63)] ? W[i] : 0.f);
}

// WO: [DM,DM]; mask = g[e%64, d%64]
__global__ void mask_wo_kernel(const float* __restrict__ W, const int* __restrict__ g,
                               half_t* __restrict__ d) {
  int i = blockIdx.x * blockDim.x + threadIdx.x;
  if (i >= kDM * kDM) return;
  int dc = i & (kDM - 1);
  int e  = i >> 9;
  d[i] = (half_t)(g[((e & 63) << 6) + (dc & 63)] ? W[i] : 0.f);
}

// L2-normalize chunks of length `chunk` within rows of a [rows, cols] f16 array
__global__ void l2norm_kernel(half_t* __restrict__ data, int rows, int cols, int chunk) {
  int nchunks = cols / chunk;
  int idx = blockIdx.x * blockDim.x + threadIdx.x;
  if (idx >= rows * nchunks) return;
  int r = idx / nchunks, c = idx % nchunks;
  half_t* p = data + (size_t)r * cols + (size_t)c * chunk;
  float ss = 0.f;
  for (int i = 0; i < chunk; ++i) { float x = (float)p[i]; ss += x * x; }
  float inv = 1.f / fmaxf(sqrtf(ss), 1e-12f);
  for (int i = 0; i < chunk; ++i) p[i] = (half_t)((float)p[i] * inv);
}

// Pack Eq/Ek [B,H,T,64]: [0:32)=wa*QA / KA, [32:48)=wp*QP / KP, [48:64)=0
__global__ void pack_e_kernel(const half_t* __restrict__ QA, const half_t* __restrict__ KA,
                              const half_t* __restrict__ QP, const half_t* __restrict__ KP,
                              const float* __restrict__ wa, const float* __restrict__ wp,
                              half_t* __restrict__ Eq, half_t* __restrict__ Ek) {
  int idx = blockIdx.x * blockDim.x + threadIdx.x;
  if (idx >= kB * kH * kT * 64) return;
  int c = idx & 63;
  int rest = idx >> 6;
  int t  = rest & (kT - 1); rest >>= 10;
  int h_ = rest & (kH - 1);
  int b_ = rest >> 3;
  size_t tok = (size_t)b_ * kT + t;
  float q = 0.f, k = 0.f;
  if (c < 32) {
    q = wa[0] * (float)QA[tok * kDA + h_ * kDAh + c];
    k = (float)KA[tok * kDA + h_ * kDAh + c];
  } else if (c < 48) {
    int cc = c - 32;
    q = wp[0] * (float)QP[tok * kDP + h_ * kDPh + cc];
    k = (float)KP[tok * kDP + h_ * kDPh + cc];
  }
  Eq[idx] = (half_t)q;
  Ek[idx] = (half_t)k;
}

// V transpose: Vt[b,h,kc,s] = Vh[(b*T+s)*512 + h*64+kc]  (coalesced B-operand)
__global__ void transpose_v_kernel(const half_t* __restrict__ Vh, half_t* __restrict__ Vt) {
  int idx = blockIdx.x * blockDim.x + threadIdx.x;
  if (idx >= kB * kH * kDK * kT) return;
  int s  = idx & (kT - 1);
  int rest = idx >> 10;
  int kc = rest & (kDK - 1); rest >>= 6;
  int h_ = rest & (kH - 1);
  int b_ = rest >> 3;
  Vt[idx] = Vh[((size_t)b_ * kT + s) * kDM + h_ * kDK + kc];
}

// ---------------------------------------------------------------------------
// n-blocked WMMA GEMM: each wave computes a 16x64 strip of C = A[M,K]*W[N,K]^T
// (+bias, f16 out). One A fragment feeds 4 WMMAs per k-step.
// ---------------------------------------------------------------------------
__global__ void gemm_f16_wmma(const half_t* __restrict__ A, int lda,
                              const half_t* __restrict__ W, int ldb,
                              const float* __restrict__ bias,
                              half_t* __restrict__ C, int ldc,
                              int M, int N, int K) {
  int wave = threadIdx.x >> 5, lane = threadIdx.x & 31;
  int hh = lane >> 4, ln = lane & 15;
  int nGroups = N >> 6;                     // groups of 4 n-tiles
  int tile = blockIdx.x * 8 + wave;
  int mT = tile / nGroups, g = tile % nGroups;
  int m0 = mT << 4, n0 = g << 6;
  const half_t* aRow = A + (size_t)(m0 + ln) * lda;
  const half_t* bRow = W + (size_t)(n0 + ln) * ldb;
  v8f acc0 = zero8(), acc1 = zero8(), acc2 = zero8(), acc3 = zero8();
  for (int kk = 0; kk < K; kk += 32) {
    v16h av = load_tile_f16(aRow + kk, hh, true);
    v16h b0 = load_tile_f16(bRow + kk, hh, true);
    v16h b1 = load_tile_f16(bRow + (size_t)16 * ldb + kk, hh, true);
    v16h b2 = load_tile_f16(bRow + (size_t)32 * ldb + kk, hh, true);
    v16h b3 = load_tile_f16(bRow + (size_t)48 * ldb + kk, hh, true);
    acc0 = wmma_f16(av, b0, acc0);
    acc1 = wmma_f16(av, b1, acc1);
    acc2 = wmma_f16(av, b2, acc2);
    acc3 = wmma_f16(av, b3, acc3);
  }
#pragma unroll
  for (int r = 0; r < 8; ++r) {
    int m = m0 + r + (hh << 3);
    half_t* crow = C + (size_t)m * ldc + n0 + ln;
    crow[0]  = (half_t)(acc0[r] + bias[n0 + ln]);
    crow[16] = (half_t)(acc1[r] + bias[n0 + ln + 16]);
    crow[32] = (half_t)(acc2[r] + bias[n0 + ln + 32]);
    crow[48] = (half_t)(acc3[r] + bias[n0 + ln + 48]);
  }
}

// ---------------------------------------------------------------------------
// sim kernel: per 16x16 tile of sim[b,h], compute a 32x32 block of S = Q K^T
// (4 WMMA tiles, zero-padded at the T boundary), band-average 17 diagonals via
// LDS, add the aux/pos gram tile (Eq Ek^T, wa/wp pre-folded), scale tn by w,
// write f32 straight into the attn region of d_out in [b,t,h,s] layout.
// ---------------------------------------------------------------------------
__global__ void sim_wmma(const half_t* __restrict__ Qh, const half_t* __restrict__ Kh,
                         const half_t* __restrict__ Eq, const half_t* __restrict__ Ek,
                         const float* __restrict__ wptr, float* __restrict__ attn) {
  __shared__ float Sblk[8][32][33];
  int wave = threadIdx.x >> 5, lane = threadIdx.x & 31;
  int hh = lane >> 4, ln = lane & 15;
  int tile = blockIdx.x * 8 + wave;        // [0, 64*64*64)
  int bTile = tile & 63; tile >>= 6;
  int aTile = tile & 63; tile >>= 6;
  int bh = tile;                            // b*H + h
  int b_ = bh >> 3, h_ = bh & 7;
  int a0 = aTile << 4, b0 = bTile << 4;

  const half_t* Qb = Qh + (size_t)b_ * kT * kDM + (size_t)h_ * kDK;  // row stride 512
  const half_t* Kb = Kh + (size_t)b_ * kT * kDM + (size_t)h_ * kDK;
  float w0 = wptr[0];

  // Aux+pos gram tile (K = 64, 48 used)
  v8f sap = zero8();
  {
    const half_t* eq = Eq + ((size_t)bh * kT + a0 + ln) * 64;
    const half_t* ek = Ek + ((size_t)bh * kT + b0 + ln) * 64;
#pragma unroll
    for (int kk = 0; kk < 64; kk += 32)
      sap = wmma_f16(load_tile_f16(eq + kk, hh, true), load_tile_f16(ek + kk, hh, true), sap);
  }

  // 32x32 S block at (a0-8, b0-8)
#pragma unroll
  for (int rr = 0; rr < 2; ++rr) {
#pragma unroll
    for (int cc = 0; cc < 2; ++cc) {
      int r0 = a0 - kM1 + (rr << 4);
      int c0 = b0 - kM1 + (cc << 4);
      int qr = r0 + ln;  bool qok = (qr >= 0) && (qr < kT);
      int kr = c0 + ln;  bool kok = (kr >= 0) && (kr < kT);
      const half_t* qp = Qb + (size_t)(qok ? qr : 0) * kDM;
      const half_t* kp = Kb + (size_t)(kok ? kr : 0) * kDM;
      v8f s = zero8();
#pragma unroll
      for (int kk = 0; kk < kDK; kk += 32)
        s = wmma_f16(load_tile_f16(qp + kk, hh, qok), load_tile_f16(kp + kk, hh, kok), s);
#pragma unroll
      for (int r = 0; r < 8; ++r)
        Sblk[wave][(rr << 4) + r + (hh << 3)][(cc << 4) + ln] = s[r];
    }
  }
  __syncthreads();

#pragma unroll
  for (int r = 0; r < 8; ++r) {
    int m = r + (hh << 3);
    int n = ln;
    int a = a0 + m, b = b0 + n;
    float num = 0.f;
#pragma unroll
    for (int u = 0; u <= kM1 + kM2; ++u) num += Sblk[wave][m + u][n + u];
    int mn = a < b ? a : b;
    int mx = a < b ? b : a;
    int c1 = mn < kM1 ? mn : kM1;
    int c2 = (kT - 1 - mx) < kM2 ? (kT - 1 - mx) : kM2;
    float cnt = (float)(c1 + c2 + 1);
    attn[(((size_t)b_ * kT + a) * kH + h_) * kT + b] = w0 * num / cnt + sap[r];
  }
}

// ---------------------------------------------------------------------------
// Row softmax in place over length-1024 rows (B*T*H rows)
// ---------------------------------------------------------------------------
__global__ void softmax_rows(float* __restrict__ data) {
  __shared__ float red[256];
  float* p = data + (size_t)blockIdx.x * kT;
  int tid = threadIdx.x;
  float v[4], mx = -3.4e38f;
#pragma unroll
  for (int i = 0; i < 4; ++i) { v[i] = p[tid + 256 * i]; mx = fmaxf(mx, v[i]); }
  red[tid] = mx; __syncthreads();
  for (int s = 128; s > 0; s >>= 1) {
    if (tid < s) red[tid] = fmaxf(red[tid], red[tid + s]);
    __syncthreads();
  }
  mx = red[0]; __syncthreads();
  float sum = 0.f;
#pragma unroll
  for (int i = 0; i < 4; ++i) { v[i] = __expf(v[i] - mx); sum += v[i]; }
  red[tid] = sum; __syncthreads();
  for (int s = 128; s > 0; s >>= 1) {
    if (tid < s) red[tid] += red[tid + s];
    __syncthreads();
  }
  float inv = 1.f / red[0];
#pragma unroll
  for (int i = 0; i < 4; ++i) p[tid + 256 * i] = v[i] * inv;
}

// ---------------------------------------------------------------------------
// deta = attn @ V per (b,h). Each wave computes a full 16x64 strip (all of
// D_K) so the 256MB f32 attn stream is read exactly once; one A fragment
// (f32->f16 cvt on load) feeds 4 WMMAs per k-step. Vt gives contiguous
// B-operand loads. Writes deta f32 to d_out and an f16 copy for outproj.
// ---------------------------------------------------------------------------
__global__ void attnv_wmma(const float* __restrict__ attn, const half_t* __restrict__ Vt,
                           float* __restrict__ deta, half_t* __restrict__ detah) {
  int wave = threadIdx.x >> 5, lane = threadIdx.x & 31;
  int hh = lane >> 4, ln = lane & 15;
  int tile = blockIdx.x * 8 + wave;        // [0, 64*64)
  int tT = tile & 63;
  int bh = tile >> 6;
  int b_ = bh >> 3, h_ = bh & 7;
  int t0 = tT << 4;

  const float*  aBase = attn + (((size_t)b_ * kT + t0 + ln) * kH + h_) * kT;
  const half_t* vBase = Vt + ((size_t)bh * kDK + ln) * kT;   // rows kc, stride kT

  v8f acc0 = zero8(), acc1 = zero8(), acc2 = zero8(), acc3 = zero8();
  for (int kk = 0; kk < kT; kk += 32) {
    v16h av = load_tile_f32cvt(aBase + kk, hh);
    v16h b0 = load_tile_f16(vBase + kk, hh, true);
    v16h b1 = load_tile_f16(vBase + (size_t)16 * kT + kk, hh, true);
    v16h b2 = load_tile_f16(vBase + (size_t)32 * kT + kk, hh, true);
    v16h b3 = load_tile_f16(vBase + (size_t)48 * kT + kk, hh, true);
    acc0 = wmma_f16(av, b0, acc0);
    acc1 = wmma_f16(av, b1, acc1);
    acc2 = wmma_f16(av, b2, acc2);
    acc3 = wmma_f16(av, b3, acc3);
  }
#pragma unroll
  for (int r = 0; r < 8; ++r) {
    int t = t0 + r + (hh << 3);
    size_t oi = ((size_t)b_ * kT + t) * kDM + h_ * kDK + ln;
    float v0 = acc0[r], v1 = acc1[r], v2 = acc2[r], v3 = acc3[r];
    deta[oi]       = v0;  detah[oi]      = (half_t)v0;
    deta[oi + 16]  = v1;  detah[oi + 16] = (half_t)v1;
    deta[oi + 32]  = v2;  detah[oi + 32] = (half_t)v2;
    deta[oi + 48]  = v3;  detah[oi + 48] = (half_t)v3;
  }
}

// ---------------------------------------------------------------------------
// out = x + deta @ (WO*mask)^T + bO   (M=8192, N=K=512), 16x64 per wave
// ---------------------------------------------------------------------------
__global__ void outproj_wmma(const half_t* __restrict__ deta, const half_t* __restrict__ WOm,
                             const float* __restrict__ x, const float* __restrict__ bO,
                             float* __restrict__ out) {
  int wave = threadIdx.x >> 5, lane = threadIdx.x & 31;
  int hh = lane >> 4, ln = lane & 15;
  int tile = blockIdx.x * 8 + wave;        // [0, 512*8)
  int mT = tile >> 3, g = tile & 7;
  int m0 = mT << 4, n0 = g << 6;
  const half_t* aRow = deta + (size_t)(m0 + ln) * kDM;
  const half_t* bRow = WOm + (size_t)(n0 + ln) * kDM;
  v8f acc0 = zero8(), acc1 = zero8(), acc2 = zero8(), acc3 = zero8();
  for (int kk = 0; kk < kDM; kk += 32) {
    v16h av = load_tile_f16(aRow + kk, hh, true);
    v16h b0 = load_tile_f16(bRow + kk, hh, true);
    v16h b1 = load_tile_f16(bRow + (size_t)16 * kDM + kk, hh, true);
    v16h b2 = load_tile_f16(bRow + (size_t)32 * kDM + kk, hh, true);
    v16h b3 = load_tile_f16(bRow + (size_t)48 * kDM + kk, hh, true);
    acc0 = wmma_f16(av, b0, acc0);
    acc1 = wmma_f16(av, b1, acc1);
    acc2 = wmma_f16(av, b2, acc2);
    acc3 = wmma_f16(av, b3, acc3);
  }
#pragma unroll
  for (int r = 0; r < 8; ++r) {
    int m = m0 + r + (hh << 3);
    size_t oi = (size_t)m * kDM + n0 + ln;
    out[oi]      = x[oi]      + acc0[r] + bO[n0 + ln];
    out[oi + 16] = x[oi + 16] + acc1[r] + bO[n0 + ln + 16];
    out[oi + 32] = x[oi + 32] + acc2[r] + bO[n0 + ln + 32];
    out[oi + 48] = x[oi + 48] + acc3[r] + bO[n0 + ln + 48];
  }
}

// ---------------------------------------------------------------------------
extern "C" void kernel_launch(void* const* d_in, const int* in_sizes, int n_in,
                              void* d_out, int out_size, void* d_ws, size_t ws_size,
                              hipStream_t stream) {
  (void)in_sizes; (void)n_in; (void)out_size; (void)ws_size;
  const float* x    = (const float*)d_in[0];
  const float* aux  = (const float*)d_in[1];
  const float* pos  = (const float*)d_in[2];
  const float* Wq   = (const float*)d_in[3];
  const float* bq   = (const float*)d_in[4];
  const float* Wk   = (const float*)d_in[5];
  const float* bk   = (const float*)d_in[6];
  const float* Wv   = (const float*)d_in[7];
  const float* bv   = (const float*)d_in[8];
  const float* Wqa  = (const float*)d_in[9];
  const float* bqa  = (const float*)d_in[10];
  const float* Wka  = (const float*)d_in[11];
  const float* bka  = (const float*)d_in[12];
  const float* Wqp  = (const float*)d_in[13];
  const float* bqp  = (const float*)d_in[14];
  const float* Wkp  = (const float*)d_in[15];
  const float* bkp  = (const float*)d_in[16];
  const float* WO   = (const float*)d_in[17];
  const float* bO   = (const float*)d_in[18];
  const float* w    = (const float*)d_in[19];
  const float* wa   = (const float*)d_in[20];
  const float* wp   = (const float*)d_in[21];
  const int*   gdep = (const int*)d_in[22];

  float* out_base  = (float*)d_out;                              // [B,T,512]
  float* attn_base = out_base + (size_t)kB * kT * kDM;           // [B,T,H*T]
  float* deta_base = attn_base + (size_t)kB * kT * kH * kT;      // [B,T,512]

  char* wsb = (char*)d_ws;
  auto carve = [&](size_t bytes) {
    void* p = (void*)wsb;
    wsb += (bytes + 255) & ~(size_t)255;
    return p;
  };
  half_t* xh    = (half_t*)carve((size_t)kNT * kDM * 2);
  half_t* auxh  = (half_t*)carve((size_t)kNT * kDA * 2);
  half_t* posh  = (half_t*)carve((size_t)kNT * kDP * 2);
  half_t* Wqm   = (half_t*)carve((size_t)kDM * kDM * 2);
  half_t* Wkm   = (half_t*)carve((size_t)kDM * kDM * 2);
  half_t* Wvm   = (half_t*)carve((size_t)kDM * kDM * 2);
  half_t* Wqah  = (half_t*)carve((size_t)kDA * kDA * 2);
  half_t* Wkah  = (half_t*)carve((size_t)kDA * kDA * 2);
  half_t* Wqph  = (half_t*)carve((size_t)kDP * kDP * 2);
  half_t* Wkph  = (half_t*)carve((size_t)kDP * kDP * 2);
  half_t* WOm   = (half_t*)carve((size_t)kDM * kDM * 2);
  half_t* Qh    = (half_t*)carve((size_t)kNT * kDM * 2);
  half_t* Kh    = (half_t*)carve((size_t)kNT * kDM * 2);
  half_t* Vh    = (half_t*)carve((size_t)kNT * kDM * 2);
  half_t* Vt    = (half_t*)carve((size_t)kNT * kDM * 2);
  half_t* QAh   = (half_t*)carve((size_t)kNT * kDA * 2);
  half_t* KAh   = (half_t*)carve((size_t)kNT * kDA * 2);
  half_t* QPh   = (half_t*)carve((size_t)kNT * kDP * 2);
  half_t* KPh   = (half_t*)carve((size_t)kNT * kDP * 2);
  half_t* Eqh   = (half_t*)carve((size_t)kB * kH * kT * 64 * 2);
  half_t* Ekh   = (half_t*)carve((size_t)kB * kH * kT * 64 * 2);
  half_t* detah = (half_t*)carve((size_t)kNT * kDM * 2);

  const int TB = 256;
  auto blocks = [](int n, int tb) { return (n + tb - 1) / tb; };

  // 1) convert inputs / small weights to f16
  cvt_f16_kernel<<<blocks(kNT * kDM, TB), TB, 0, stream>>>(x, xh, kNT * kDM);
  cvt_f16_kernel<<<blocks(kNT * kDA, TB), TB, 0, stream>>>(aux, auxh, kNT * kDA);
  cvt_f16_kernel<<<blocks(kNT * kDP, TB), TB, 0, stream>>>(pos, posh, kNT * kDP);
  cvt_f16_kernel<<<blocks(kDA * kDA, TB), TB, 0, stream>>>(Wqa, Wqah, kDA * kDA);
  cvt_f16_kernel<<<blocks(kDA * kDA, TB), TB, 0, stream>>>(Wka, Wkah, kDA * kDA);
  cvt_f16_kernel<<<blocks(kDP * kDP, TB), TB, 0, stream>>>(Wqp, Wqph, kDP * kDP);
  cvt_f16_kernel<<<blocks(kDP * kDP, TB), TB, 0, stream>>>(Wkp, Wkph, kDP * kDP);

  // 2) masked weights
  mask_qkv_kernel<<<blocks(kDM * kDM, TB), TB, 0, stream>>>(Wq, gdep, Wqm);
  mask_qkv_kernel<<<blocks(kDM * kDM, TB), TB, 0, stream>>>(Wk, gdep, Wkm);
  mask_qkv_kernel<<<blocks(kDM * kDM, TB), TB, 0, stream>>>(Wv, gdep, Wvm);
  mask_wo_kernel<<<blocks(kDM * kDM, TB), TB, 0, stream>>>(WO, gdep, WOm);

  // 3) projection GEMMs (+bias), f16 out; 16x64 strips per wave
  gemm_f16_wmma<<<(kNT/16)*(kDM/64)/8, TB, 0, stream>>>(xh, kDM, Wqm, kDM, bq, Qh, kDM, kNT, kDM, kDM);
  gemm_f16_wmma<<<(kNT/16)*(kDM/64)/8, TB, 0, stream>>>(xh, kDM, Wkm, kDM, bk, Kh, kDM, kNT, kDM, kDM);
  gemm_f16_wmma<<<(kNT/16)*(kDM/64)/8, TB, 0, stream>>>(xh, kDM, Wvm, kDM, bv, Vh, kDM, kNT, kDM, kDM);
  gemm_f16_wmma<<<(kNT/16)*(kDA/64)/8, TB, 0, stream>>>(auxh, kDA, Wqah, kDA, bqa, QAh, kDA, kNT, kDA, kDA);
  gemm_f16_wmma<<<(kNT/16)*(kDA/64)/8, TB, 0, stream>>>(auxh, kDA, Wkah, kDA, bka, KAh, kDA, kNT, kDA, kDA);
  gemm_f16_wmma<<<(kNT/16)*(kDP/64)/8, TB, 0, stream>>>(posh, kDP, Wqph, kDP, bqp, QPh, kDP, kNT, kDP, kDP);
  gemm_f16_wmma<<<(kNT/16)*(kDP/64)/8, TB, 0, stream>>>(posh, kDP, Wkph, kDP, bkp, KPh, kDP, kNT, kDP, kDP);

  // 4) l2 normalization (per-head chunks)
  l2norm_kernel<<<blocks(kNT * kH, TB), TB, 0, stream>>>(Qh, kNT, kDM, kDK);
  l2norm_kernel<<<blocks(kNT * kH, TB), TB, 0, stream>>>(Kh, kNT, kDM, kDK);
  l2norm_kernel<<<blocks(kNT * kH, TB), TB, 0, stream>>>(QAh, kNT, kDA, kDAh);
  l2norm_kernel<<<blocks(kNT * kH, TB), TB, 0, stream>>>(KAh, kNT, kDA, kDAh);
  l2norm_kernel<<<blocks(kNT * kH, TB), TB, 0, stream>>>(QPh, kNT, kDP, kDPh);
  l2norm_kernel<<<blocks(kNT * kH, TB), TB, 0, stream>>>(KPh, kNT, kDP, kDPh);

  // 5) pack aux+pos features (wa/wp folded into Eq); transpose V
  pack_e_kernel<<<blocks(kB * kH * kT * 64, TB), TB, 0, stream>>>(QAh, KAh, QPh, KPh, wa, wp, Eqh, Ekh);
  transpose_v_kernel<<<blocks(kB * kH * kDK * kT, TB), TB, 0, stream>>>(Vh, Vt);

  // 6) sim = w*tn(QK^T) + Eq Ek^T, written into attn region of d_out
  sim_wmma<<<(kB * kH * (kT/16) * (kT/16)) / 8, TB, 0, stream>>>(Qh, Kh, Eqh, Ekh, w, attn_base);

  // 7) softmax in place (B*T*H rows of length T)
  softmax_rows<<<kB * kT * kH, TB, 0, stream>>>(attn_base);

  // 8) deta = attn @ V (f32 out + f16 copy); attn read exactly once
  attnv_wmma<<<(kB * kH * (kT/16)) / 8, TB, 0, stream>>>(attn_base, Vt, deta_base, detah);

  // 9) out = x + deta @ WOm^T + bO
  outproj_wmma<<<(kNT/16)*(kDM/64)/8, TB, 0, stream>>>(detah, WOm, x, bO, out_base);
}
